// STAttentionBlock_40681930227721
// MI455X (gfx1250) — matrix operand order repaired and checked by reading, not compiled, not run on hardware
//
#include <hip/hip_runtime.h>
#include <hip/hip_bf16.h>

// ============================================================================
// ST attention block for MI455X (gfx1250, wave32, WMMA).
// conv1x1 -> batched GEMM via v_wmma_f32_16x16x32_f16 (f16 in, f32 acc),
// double-buffered fragment-major LDS tiles, register-pipelined staging,
// fused bias/bn/residual/lrelu epilogues.
// Small attention einsums -> LDS-tiled VALU kernels.
// ============================================================================

typedef __attribute__((ext_vector_type(16))) _Float16 v16h;
typedef __attribute__((ext_vector_type(8)))  _Float16 v8h;
typedef __attribute__((ext_vector_type(8)))  float    v8f;

#define NB    32
#define CINC  128
#define COUTC 256
#define TDIM  64
#define VDIM  25
#define SDIM  8
#define ICD   32
#define WDIM  3
#define TV    1600   // T*V
#define TWV   4800   // T*W*V

// 1/sqrt(1 + 1e-5)
#define BN_SCALE 0.9999950000374997f

// LDS row stride (halves): 80B rows keep 16B-aligned b128 chunks and give
// conflict-free fragment reads (20*c mod 64 distinct over c=0..15).
#define LSTR 40

// ---------------------------------------------------------------------------
__device__ __forceinline__ float pe_val(int c, int pos, int C) {
  float j   = (float)(c >> 1);
  float dv  = __expf(-2.f * j * 9.210340371976184f / (float)C);
  float ang = (float)pos * dv;
  return (c & 1) ? __cosf(ang) : __sinf(ang);
}

// ---------------------------------------------------------------------------
// y = x + pos_enc(CIN, T, V, 'spatial')   (pos = v)
// ---------------------------------------------------------------------------
__global__ void k_pe_spatial(const float* __restrict__ x, float* __restrict__ y,
                             long total) {
  long i = (long)blockIdx.x * blockDim.x + threadIdx.x;
  if (i >= total) return;
  int v = (int)(i % VDIM);
  int c = (int)((i / TV) % CINC);
  y[i] = x[i] + pe_val(c, v, CINC);
}

// ---------------------------------------------------------------------------
// z = y2 + pos_enc(COUT, T, V, 'temporal')   (pos = t)
// ---------------------------------------------------------------------------
__global__ void k_pe_temporal(const float* __restrict__ y2, float* __restrict__ z,
                              long total) {
  long i = (long)blockIdx.x * blockDim.x + threadIdx.x;
  if (i >= total) return;
  int t = (int)((i / VDIM) % TDIM);
  int c = (int)((i / TV) % COUTC);
  z[i] = y2[i] + pe_val(c, t, COUTC);
}

// ---------------------------------------------------------------------------
// up[n,c,t,w*V+v] = y[n,c,t+w-1,v]  (zero-padded in t)
// ---------------------------------------------------------------------------
__global__ void k_unfold(const float* __restrict__ y, float* __restrict__ up,
                         long total) {
  long i = (long)blockIdx.x * blockDim.x + threadIdx.x;
  if (i >= total) return;
  int  u   = (int)(i % TWV);
  long nct = i / TWV;             // (n*CIN + c)*T + t
  int  t   = (int)(nct % TDIM);
  long nc  = nct / TDIM;
  int  w   = u / VDIM, v = u % VDIM;
  int  ts  = t + w - 1;
  up[i] = (ts >= 0 && ts < TDIM) ? y[(nc * TDIM + ts) * VDIM + v] : 0.f;
}

// ---------------------------------------------------------------------------
// Batched WMMA GEMM:  C[b] = epi( A(MxK) * B[b](KxN) )
//   A row-major (weights, shared over batch), B/C row-major per batch.
//   M % 64 == 0, N % 64 == 0, K % 32 == 0.
//   epi: +bias[row]; optional bn (g,beta); optional +res; optional lrelu(0.1)
// grid = (N/64, M/64, batch), block = 128 (4 waves, wave32).
// Wave (2x2) computes a 32x32 sub-tile: 2 A frags x 2 B frags, 4 WMMAs/K-step.
// LDS tiles fragment-major and DOUBLE-BUFFERED; staging is register-pipelined
// (unrolled vector global loads for the next tile issued before this tile's
// WMMAs), one barrier per K-step.
//   A stored [row][k], k contiguous; lane half h takes K16h..16h+15 -> lane-half reads
//   16 contiguous halves (2x ds_load_b128).  B transposed [col][k].
// ---------------------------------------------------------------------------
__global__ __launch_bounds__(128)
void k_gemm_wmma(const float* __restrict__ A, int K,
                 const float* __restrict__ B, long strideB, int Ncols,
                 float* __restrict__ C, long strideC,
                 const float* __restrict__ bias,
                 const float* __restrict__ gamma, const float* __restrict__ beta,
                 const float* __restrict__ res, long strideRes, int do_lrelu) {
  __shared__ _Float16 Als[2][64][LSTR];   // [buf][row][k]
  __shared__ _Float16 Bls[2][64][LSTR];   // [buf][col][k]

  const int tid  = threadIdx.x;
  const int wave = tid >> 5;
  const int lane = tid & 31;
  const int mh   = (wave >> 1) * 32;    // wave row-half within 64
  const int nhf  = (wave & 1) * 32;     // wave col-half within 64
  const int half = lane >> 4;           // lane half selects K chunk
  const int l15  = lane & 15;
  const int tn   = blockIdx.x * 64;
  const int tm   = blockIdx.y * 64;
  const float* Bb = B + (long)blockIdx.z * strideB;
  float*       Cb = C + (long)blockIdx.z * strideC;

  // staging assignment:
  //  A: row ar = tid>>1, k-halve akh = (tid&1)*16 (16 contiguous floats)
  //     contiguous store offsets: akh==0 -> {0,8}; akh==16 -> {16,24}
  //  B: col bc = tid&63, k-half bkq = (tid>>6)*16 (16 K-strided floats)
  const int ar    = tid >> 1;
  const int akh   = (tid & 1) * 16;
  const int apos0 = (tid & 1) ? 16 : 0;
  const int bc    = tid & 63;
  const int bkq   = (tid >> 6) * 16;

  const float* Aptr = A + (long)(tm + ar) * K + akh;
  const float* Bptr = Bb + (long)bkq * Ncols + tn + bc;

  float ra[16], rb[16];

  auto load_regs = [&](int it) {
    const float* Ap = Aptr + it * 32;
    const float* Bp = Bptr + (long)it * 32 * Ncols;
#pragma unroll
    for (int j = 0; j < 16; ++j) ra[j] = Ap[j];
#pragma unroll
    for (int j = 0; j < 16; ++j) rb[j] = Bp[(long)j * Ncols];
  };
  auto store_tiles = [&](int buf) {
    v8h pa0, pa1, pb0, pb1;
#pragma unroll
    for (int j = 0; j < 8; ++j) {
      pa0[j] = (_Float16)ra[j];
      pa1[j] = (_Float16)ra[8 + j];
      pb0[j] = (_Float16)rb[j];
      pb1[j] = (_Float16)rb[8 + j];
    }
    *(v8h*)&Als[buf][ar][apos0]      = pa0;
    *(v8h*)&Als[buf][ar][apos0 + 8]  = pa1;
    *(v8h*)&Bls[buf][bc][bkq]        = pb0;
    *(v8h*)&Bls[buf][bc][bkq + 8]    = pb1;
  };

  v8f acc[2][2] = {{v8f{}, v8f{}}, {v8f{}, v8f{}}};

  // prologue: stage tile 0
  load_regs(0);
  store_tiles(0);
  __syncthreads();

  const int nK = K >> 5;
  for (int it = 0; it < nK; ++it) {
    const int cur = it & 1;
    if (it + 1 < nK) load_regs(it + 1);              // pipeline next tile
    if (it + 2 < nK) {                               // warm L2 two tiles out
      __builtin_prefetch(Aptr + (it + 2) * 32, 0, 1);
      __builtin_prefetch(Bptr + (long)(it + 2) * 32 * Ncols, 0, 1);
    }

    // fragments: two 16B LDS vector loads each
    v16h af[2], bf[2];
#pragma unroll
    for (int mt = 0; mt < 2; ++mt) {
      const _Float16* p = &Als[cur][mh + mt * 16 + l15][half * 16];
      v8h a0 = *(const v8h*)p;
      v8h a1 = *(const v8h*)(p + 8);
      af[mt] = __builtin_shufflevector(a0, a1, 0, 1, 2, 3, 4, 5, 6, 7,
                                       8, 9, 10, 11, 12, 13, 14, 15);
    }
#pragma unroll
    for (int nt = 0; nt < 2; ++nt) {
      const _Float16* p = &Bls[cur][nhf + nt * 16 + l15][half * 16];
      v8h b0 = *(const v8h*)p;
      v8h b1 = *(const v8h*)(p + 8);
      bf[nt] = __builtin_shufflevector(b0, b1, 0, 1, 2, 3, 4, 5, 6, 7,
                                       8, 9, 10, 11, 12, 13, 14, 15);
    }
#pragma unroll
    for (int mt = 0; mt < 2; ++mt)
#pragma unroll
      for (int nt = 0; nt < 2; ++nt)
        acc[mt][nt] = __builtin_amdgcn_wmma_f32_16x16x32_f16(
            false, af[mt], false, bf[nt], (short)0, acc[mt][nt], false, false);

    if (it + 1 < nK) store_tiles(1 - cur);           // fill other buffer
    __syncthreads();
  }

  // epilogue; D layout: VGPR r -> row (r + 8*half), col = lane&15
#pragma unroll
  for (int mt = 0; mt < 2; ++mt) {
    int rbase = tm + mh + mt * 16 + (half << 3);
#pragma unroll
    for (int nt = 0; nt < 2; ++nt) {
      int col = tn + nhf + nt * 16 + l15;
#pragma unroll
      for (int r = 0; r < 8; ++r) {
        int   row = rbase + r;
        float v   = acc[mt][nt][r] + bias[row];
        if (gamma) v = v * (gamma[row] * BN_SCALE) + beta[row];
        if (res)   v += res[(long)blockIdx.z * strideRes + (long)row * Ncols + col];
        if (do_lrelu) v = v > 0.f ? v : 0.1f * v;
        Cb[(long)row * Ncols + col] = v;
      }
    }
  }
}

// ---------------------------------------------------------------------------
// ya[n,c,t,v] = sum_u y[n,c,t,u] * graph_a[u,v]
// ---------------------------------------------------------------------------
__global__ void k_ya(const float* __restrict__ y, const float* __restrict__ ga,
                     float* __restrict__ ya, long total) {
  long i = (long)blockIdx.x * blockDim.x + threadIdx.x;
  if (i >= total) return;
  int  v    = (int)(i % VDIM);
  long base = (i / VDIM) * VDIM;
  float acc = 0.f;
#pragma unroll
  for (int u = 0; u < VDIM; ++u) acc += y[base + u] * ga[u * VDIM + v];
  ya[i] = acc;
}

// ---------------------------------------------------------------------------
// d[n,s,w] = sigmoid( mean_{cc,t,v} ( dd[n,s*16+cc,t,v] - up[n,s*16+cc,t,w*V+v] ) )
// one block per (n,s,w)
// ---------------------------------------------------------------------------
__global__ __launch_bounds__(256)
void k_dred(const float* __restrict__ dd, const float* __restrict__ up,
            float* __restrict__ d) {
  int b = blockIdx.x;
  int w = b % WDIM, s = (b / WDIM) % SDIM, n = b / (WDIM * SDIM);
  float part = 0.f;
  for (int i = threadIdx.x; i < 16 * TDIM * VDIM; i += 256) {
    int  v  = i % VDIM;
    int  t  = (i / VDIM) % TDIM;
    int  cc = i / (VDIM * TDIM);
    long ch = (long)n * CINC + s * 16 + cc;
    part += dd[(ch * TDIM + t) * VDIM + v]
          - up[(ch * TDIM + t) * TWV + w * VDIM + v];
  }
  __shared__ float red[256];
  red[threadIdx.x] = part;
  __syncthreads();
  for (int off = 128; off > 0; off >>= 1) {
    if (threadIdx.x < off) red[threadIdx.x] += red[threadIdx.x + off];
    __syncthreads();
  }
  if (threadIdx.x == 0) {
    float m = red[0] / (float)(16 * TDIM * VDIM);
    d[b] = 1.f / (1.f + __expf(-m));
  }
}

// ---------------------------------------------------------------------------
// att[n,s,u,v] = (1/(IC*T)) * sum_{c,t} q[n,s,c,t,u] * k[n,s,c,t,v]
// one block per (n,s); channel chunks of 2 staged through LDS
// ---------------------------------------------------------------------------
__global__ __launch_bounds__(256)
void k_att(const float* __restrict__ q, const float* __restrict__ k,
           float* __restrict__ att) {
  __shared__ float qs[2 * TDIM * 75];    // 9600 floats
  __shared__ float ks[2 * TDIM * VDIM];  // 3200 floats
  int ns = blockIdx.x, n = ns >> 3, s = ns & 7;
  const float* qb = q + ((long)(n * (SDIM * ICD) + s * ICD)) * TDIM * 75;
  const float* kb = k + ((long)(n * (SDIM * ICD) + s * ICD)) * TDIM * VDIM;
  float acc[8] = {0, 0, 0, 0, 0, 0, 0, 0};
  for (int c0 = 0; c0 < ICD; c0 += 2) {
    for (int i = threadIdx.x; i < 2 * TDIM * 75; i += 256)
      qs[i] = qb[(long)c0 * TDIM * 75 + i];
    for (int i = threadIdx.x; i < 2 * TDIM * VDIM; i += 256)
      ks[i] = kb[(long)c0 * TDIM * VDIM + i];
    __syncthreads();
    for (int o = 0; o < 8; ++o) {
      int idx = o * 256 + threadIdx.x;
      if (idx < 75 * VDIM) {
        int u = idx / VDIM, v = idx % VDIM;
        float a = acc[o];
        for (int ct = 0; ct < 2 * TDIM; ++ct)
          a += qs[ct * 75 + u] * ks[ct * VDIM + v];
        acc[o] = a;
      }
    }
    __syncthreads();
  }
  const float sc = 1.f / (float)(ICD * TDIM);
  for (int o = 0; o < 8; ++o) {
    int idx = o * 256 + threadIdx.x;
    if (idx < 75 * VDIM) att[(long)ns * 75 * VDIM + idx] = acc[o] * sc;
  }
}

// ---------------------------------------------------------------------------
// in-place: att = softmax(where(mask, att, -9e15)) * d[n,s,u%3] + att0s[s,u,v]
// mask[u,v] = graph[s, u%V, v] > 0.  one block per (n,s), one thread per row u
// ---------------------------------------------------------------------------
__global__ void k_attmix(float* __restrict__ att, const float* __restrict__ graph,
                         const float* __restrict__ d, const float* __restrict__ att0s) {
  int ns = blockIdx.x, s = ns & 7;
  int u = threadIdx.x;
  if (u >= 75) return;
  float*       row = att + ((long)ns * 75 + u) * VDIM;
  const float* g   = graph + (long)s * VDIM * VDIM + (u % VDIM) * VDIM;
  float vals[VDIM];
  float m = -3.0e38f;
#pragma unroll
  for (int v = 0; v < VDIM; ++v) {
    float val = (g[v] > 0.f) ? row[v] : -9.0e15f;
    vals[v] = val;
    if (val > m) m = val;
  }
  float sum = 0.f;
#pragma unroll
  for (int v = 0; v < VDIM; ++v) { vals[v] = __expf(vals[v] - m); sum += vals[v]; }
  float inv = 1.f / sum;
  float sc  = d[ns * WDIM + (u % WDIM)];
  const float* a0 = att0s + ((long)s * 75 + u) * VDIM;
#pragma unroll
  for (int v = 0; v < VDIM; ++v) row[v] = vals[v] * inv * sc + a0[v];
}

// ---------------------------------------------------------------------------
// y2a[n, s*128+c, t, v] = sum_u up[n,c,t,u] * att[n,s,u,v]
// one block per (t, s, n)
// ---------------------------------------------------------------------------
__global__ __launch_bounds__(256)
void k_y2a(const float* __restrict__ up, const float* __restrict__ att,
           float* __restrict__ y2a) {
  __shared__ float attl[75 * VDIM];  // 1875
  __shared__ float upl[CINC * 75];   // 9600
  int t = blockIdx.x, s = blockIdx.y, n = blockIdx.z;
  const float* ab = att + ((long)(n * SDIM + s)) * 75 * VDIM;
  for (int i = threadIdx.x; i < 75 * VDIM; i += 256) attl[i] = ab[i];
  for (int i = threadIdx.x; i < CINC * 75; i += 256) {
    int c = i / 75, u = i % 75;
    upl[i] = up[(((long)n * CINC + c) * TDIM + t) * 75 + u];
  }
  __syncthreads();
  for (int o = 0; o < 13; ++o) {
    int idx = o * 256 + threadIdx.x;
    if (idx < CINC * VDIM) {
      int c = idx / VDIM, v = idx % VDIM;
      float a = 0.f;
      for (int u = 0; u < 75; ++u) a += upl[c * 75 + u] * attl[u * VDIM + v];
      y2a[(((long)n * (SDIM * CINC) + s * CINC + c) * TDIM + t) * VDIM + v] = a;
    }
  }
}

// ---------------------------------------------------------------------------
// attt[n,s,t,q] = tanh( sum_{c,v} qt[n,s,c,t,v]*kt[n,s,c,q,v] / (IC*V) )*alphat[s]
//                 + att0t[s,t,q]
// one block per (n,s); channel chunks of 4 in LDS
// ---------------------------------------------------------------------------
__global__ __launch_bounds__(256)
void k_attt(const float* __restrict__ qk, const float* __restrict__ alphat,
            const float* __restrict__ att0t, float* __restrict__ attt) {
  __shared__ float qts[4 * TDIM * VDIM];  // 6400
  __shared__ float kts[4 * TDIM * VDIM];  // 6400
  int ns = blockIdx.x, n = ns >> 3, s = ns & 7;
  const float* qb = qk + ((long)n * (2 * SDIM * ICD) + s * ICD) * TDIM * VDIM;
  const float* kb = qk + ((long)n * (2 * SDIM * ICD) + (SDIM + s) * ICD) * TDIM * VDIM;
  float acc[16];
#pragma unroll
  for (int o = 0; o < 16; ++o) acc[o] = 0.f;
  for (int c0 = 0; c0 < ICD; c0 += 4) {
    for (int i = threadIdx.x; i < 4 * TDIM * VDIM; i += 256) {
      qts[i] = qb[(long)c0 * TDIM * VDIM + i];
      kts[i] = kb[(long)c0 * TDIM * VDIM + i];
    }
    __syncthreads();
    for (int o = 0; o < 16; ++o) {
      int   idx = o * 256 + threadIdx.x;
      int   t = idx >> 6, qq = idx & 63;
      float a = acc[o];
      for (int c = 0; c < 4; ++c)
        for (int v = 0; v < VDIM; ++v)
          a += qts[(c * TDIM + t) * VDIM + v] * kts[(c * TDIM + qq) * VDIM + v];
      acc[o] = a;
    }
    __syncthreads();
  }
  const float sc = 1.f / (float)(ICD * VDIM);
  float al = alphat[s];
  for (int o = 0; o < 16; ++o) {
    int idx = o * 256 + threadIdx.x;
    int t = idx >> 6, qq = idx & 63;
    attt[((long)ns * TDIM + t) * TDIM + qq] =
        tanhf(acc[o] * sc) * al + att0t[((long)s * TDIM + t) * TDIM + qq];
  }
}

// ---------------------------------------------------------------------------
// z2a_chunk[nn, s*256+c, q, v] = sum_t y2[n0+nn,c,t,v] * attt[n0+nn,s,t,q]
// one block per (c, s, nn)
// ---------------------------------------------------------------------------
__global__ __launch_bounds__(256)
void k_z2a(const float* __restrict__ y2, const float* __restrict__ attt,
           float* __restrict__ z2a, int n0) {
  __shared__ float attl[TDIM * TDIM];  // 4096
  __shared__ float y2l[TV];            // 1600
  int c = blockIdx.x, s = blockIdx.y, nn = blockIdx.z, n = n0 + nn;
  const float* ab = attt + ((long)(n * SDIM + s)) * TDIM * TDIM;
  for (int i = threadIdx.x; i < TDIM * TDIM; i += 256) attl[i] = ab[i];
  const float* yb = y2 + ((long)n * COUTC + c) * TV;
  for (int i = threadIdx.x; i < TV; i += 256) y2l[i] = yb[i];
  __syncthreads();
  float* ob = z2a + ((long)nn * (SDIM * COUTC) + s * COUTC + c) * TV;
  for (int o = 0; o < 7; ++o) {
    int idx = o * 256 + threadIdx.x;
    if (idx < TV) {
      int q = idx / VDIM, v = idx % VDIM;
      float a = 0.f;
      for (int t = 0; t < TDIM; ++t) a += y2l[t * VDIM + v] * attl[t * TDIM + q];
      ob[idx] = a;
    }
  }
}

// ============================================================================
// orchestration
// ============================================================================
extern "C" void kernel_launch(void* const* d_in, const int* in_sizes, int n_in,
                              void* d_out, int out_size, void* d_ws, size_t ws_size,
                              hipStream_t stream) {
  (void)in_sizes; (void)n_in; (void)out_size; (void)ws_size;
  const float* x       = (const float*)d_in[0];
  const float* graph   = (const float*)d_in[1];
  const float* ga      = (const float*)d_in[2];
  const float* w_in    = (const float*)d_in[3];
  const float* b_in    = (const float*)d_in[4];
  const float* w_inup  = (const float*)d_in[5];
  const float* b_inup  = (const float*)d_in[6];
  const float* w_diff  = (const float*)d_in[7];
  const float* b_diff  = (const float*)d_in[8];
  const float* att0s   = (const float*)d_in[9];
  const float* w_outs  = (const float*)d_in[10];
  const float* b_outs  = (const float*)d_in[11];
  const float* g_outs  = (const float*)d_in[12];
  const float* be_outs = (const float*)d_in[13];
  const float* w_ffs   = (const float*)d_in[14];
  const float* b_ffs   = (const float*)d_in[15];
  const float* g_ffs   = (const float*)d_in[16];
  const float* be_ffs  = (const float*)d_in[17];
  const float* w_ds1   = (const float*)d_in[18];
  const float* b_ds1   = (const float*)d_in[19];
  const float* g_ds1   = (const float*)d_in[20];
  const float* be_ds1  = (const float*)d_in[21];
  const float* w_ds2   = (const float*)d_in[22];
  const float* b_ds2   = (const float*)d_in[23];
  const float* g_ds2   = (const float*)d_in[24];
  const float* be_ds2  = (const float*)d_in[25];
  const float* w_int   = (const float*)d_in[26];
  const float* b_int   = (const float*)d_in[27];
  const float* alphat  = (const float*)d_in[28];
  const float* att0t   = (const float*)d_in[29];
  const float* w_outt  = (const float*)d_in[30];
  const float* b_outt  = (const float*)d_in[31];
  const float* g_outt  = (const float*)d_in[32];
  const float* be_outt = (const float*)d_in[33];
  const float* w_fft   = (const float*)d_in[34];
  const float* b_fft   = (const float*)d_in[35];
  const float* g_fft   = (const float*)d_in[36];
  const float* be_fft  = (const float*)d_in[37];
  const float* w_dt1   = (const float*)d_in[38];
  const float* b_dt1   = (const float*)d_in[39];
  const float* g_dt1   = (const float*)d_in[40];
  const float* be_dt1  = (const float*)d_in[41];
  const float* w_dt2   = (const float*)d_in[42];
  const float* b_dt2   = (const float*)d_in[43];
  const float* g_dt2   = (const float*)d_in[44];
  const float* be_dt2  = (const float*)d_in[45];
  float* out = (float*)d_out;
  float* ws  = (float*)d_ws;

  // --- workspace region map (float elems), lifetimes overlapped -------------
  // total = 93,279,744 floats (~356 MiB)
  const long oY    = 0;          // y     : 6,553,600
  const long oUP   = 6553600;    // up    : 19,660,800
  const long oK    = 26214400;   // k     : 13,107,200
  const long oQ    = 39321600;   // q     : 39,321,600
  const long oATT  = 78643200;   // att   : 480,000
  const long oATTT = 79123200;   // attt  : 1,048,576
  const long oD    = 80171776;   // d     : 768 (pad)
  const long oYA   = 80172544;   // ya    : 6,553,600
  const long oDD   = 86726144;   // dd    : 6,553,600  -> end 93,279,744
  // reuse (dead regions):
  const long oY2A  = oK;         // 52,428,800 over [k|q]
  const long oT1   = 0;          // 13,107,200 over dead [y|up)
  const long oT2   = 13107200;   // 13,107,200
  const long oY2   = 26214400;   // over dead y2a head
  const long oZ    = 39321600;
  const long oQK   = 52428800;   // 26,214,400
  const long oZ2A  = 52428800;   // chunk (8 batches) over dead qk
  const long oT3   = 0;
  const long oT4   = 13107200;

  const long szY2  = (long)COUTC * TV;  // per-batch 256x1600

  // ------------------------------ spatial stage ----------------------------
  { long tot = (long)NB * CINC * TV;
    k_pe_spatial<<<(int)((tot + 255) / 256), 256, 0, stream>>>(x, ws + oY, tot); }
  { long tot = (long)NB * CINC * TWV;
    k_unfold<<<(int)((tot + 255) / 256), 256, 0, stream>>>(ws + oY, ws + oUP, tot); }

  // k = conv1x1(y, w_in) : M=256, K=128, N=1600
  k_gemm_wmma<<<dim3(25, 4, NB), 128, 0, stream>>>(
      w_in, CINC, ws + oY, (long)CINC * TV, TV, ws + oK, (long)256 * TV,
      b_in, nullptr, nullptr, nullptr, 0, 0);
  // q = conv1x1(up, w_inup) : M=256, K=128, N=4800
  k_gemm_wmma<<<dim3(75, 4, NB), 128, 0, stream>>>(
      w_inup, CINC, ws + oUP, (long)CINC * TWV, TWV, ws + oQ, (long)256 * TWV,
      b_inup, nullptr, nullptr, nullptr, 0, 0);

  // ya = y @ graph_a ; dd = conv1x1(ya, w_diff)
  { long tot = (long)NB * CINC * TV;
    k_ya<<<(int)((tot + 255) / 256), 256, 0, stream>>>(ws + oY, ga, ws + oYA, tot); }
  k_gemm_wmma<<<dim3(25, 2, NB), 128, 0, stream>>>(
      w_diff, CINC, ws + oYA, (long)CINC * TV, TV, ws + oDD, (long)CINC * TV,
      b_diff, nullptr, nullptr, nullptr, 0, 0);
  k_dred<<<NB * SDIM * WDIM, 256, 0, stream>>>(ws + oDD, ws + oUP, ws + oD);

  // att = q.k / (IC*T) ; masked softmax * d + attention0s
  k_att<<<NB * SDIM, 256, 0, stream>>>(ws + oQ, ws + oK, ws + oATT);
  k_attmix<<<NB * SDIM, 128, 0, stream>>>(ws + oATT, graph, ws + oD, att0s);

  // y2a = einsum(up, att)  (N, 1024, T, V)
  k_y2a<<<dim3(TDIM, SDIM, NB), 256, 0, stream>>>(ws + oUP, ws + oATT, ws + oY2A);

  // t1 = bn(conv(y2a, w_outs))
  k_gemm_wmma<<<dim3(25, 4, NB), 128, 0, stream>>>(
      w_outs, SDIM * CINC, ws + oY2A, (long)SDIM * CINC * TV, TV, ws + oT1, szY2,
      b_outs, g_outs, be_outs, nullptr, 0, 0);
  // t2 = lrelu(bn(conv(x, w_ds1)) + t1)
  k_gemm_wmma<<<dim3(25, 4, NB), 128, 0, stream>>>(
      w_ds1, CINC, x, (long)CINC * TV, TV, ws + oT2, szY2,
      b_ds1, g_ds1, be_ds1, ws + oT1, szY2, 1);
  // t1 = bn(conv(t2, w_ffs))
  k_gemm_wmma<<<dim3(25, 4, NB), 128, 0, stream>>>(
      w_ffs, COUTC, ws + oT2, szY2, TV, ws + oT1, szY2,
      b_ffs, g_ffs, be_ffs, nullptr, 0, 0);
  // y2 = lrelu(bn(conv(x, w_ds2)) + t1)
  k_gemm_wmma<<<dim3(25, 4, NB), 128, 0, stream>>>(
      w_ds2, CINC, x, (long)CINC * TV, TV, ws + oY2, szY2,
      b_ds2, g_ds2, be_ds2, ws + oT1, szY2, 1);

  // ------------------------------ temporal stage ---------------------------
  { long tot = (long)NB * COUTC * TV;
    k_pe_temporal<<<(int)((tot + 255) / 256), 256, 0, stream>>>(ws + oY2, ws + oZ, tot); }
  // qk = conv(z, w_int) : M=512, K=256
  k_gemm_wmma<<<dim3(25, 8, NB), 128, 0, stream>>>(
      w_int, COUTC, ws + oZ, szY2, TV, ws + oQK, (long)512 * TV,
      b_int, nullptr, nullptr, nullptr, 0, 0);
  k_attt<<<NB * SDIM, 256, 0, stream>>>(ws + oQK, alphat, att0t, ws + oATTT);

  // z2a einsum + w_outt GEMM, chunked over batches of 8 to bound workspace
  for (int n0 = 0; n0 < NB; n0 += 8) {
    k_z2a<<<dim3(COUTC, SDIM, 8), 256, 0, stream>>>(ws + oY2, ws + oATTT, ws + oZ2A, n0);
    k_gemm_wmma<<<dim3(25, 4, 8), 128, 0, stream>>>(
        w_outt, SDIM * COUTC, ws + oZ2A, (long)SDIM * COUTC * TV, TV,
        ws + oT3 + (long)n0 * szY2, szY2, b_outt, g_outt, be_outt, nullptr, 0, 0);
  }

  // t4 = lrelu(bn(conv(y2, w_dt1)) + t3)
  k_gemm_wmma<<<dim3(25, 4, NB), 128, 0, stream>>>(
      w_dt1, COUTC, ws + oY2, szY2, TV, ws + oT4, szY2,
      b_dt1, g_dt1, be_dt1, ws + oT3, szY2, 1);
  // t3 = bn(conv(t4, w_fft))
  k_gemm_wmma<<<dim3(25, 4, NB), 128, 0, stream>>>(
      w_fft, COUTC, ws + oT4, szY2, TV, ws + oT3, szY2,
      b_fft, g_fft, be_fft, nullptr, 0, 0);
  // out = lrelu(bn(conv(y2, w_dt2)) + t3)
  k_gemm_wmma<<<dim3(25, 4, NB), 128, 0, stream>>>(
      w_dt2, COUTC, ws + oY2, szY2, TV, out, szY2,
      b_dt2, g_dt2, be_dt2, ws + oT3, szY2, 1);
}
